// TriangleAttention_33560874451052
// MI455X (gfx1250) — matrix operand order repaired and checked by reading, not compiled, not run on hardware
//
#include <hip/hip_runtime.h>
#include <hip/hip_bf16.h>
#include <stdint.h>

typedef __attribute__((ext_vector_type(16))) _Float16 v16h;
typedef __attribute__((ext_vector_type(8)))  float    v8f;

#define Nseq   256
#define Dm     128
#define Hh     4
#define DHd    32
#define NN     (Nseq*Nseq)        // 65536 positions
#define W1COLS 528                // 384 qkv + 128 gate + 4 pair + 12 pad
#define LN_EPS 1e-6f

// ---- workspace layout (bytes, all 256-aligned) ----
#define WS_W1T   ((size_t)0)                                  // [528][128] f16
#define WS_WOUTT (WS_W1T   + (size_t)W1COLS*Dm*2)             // [128][128] f16 (transposed)
#define WS_Q     (WS_WOUTT + (size_t)Dm*Dm*2)                 // [(r*4+h)*256+i][32] f16
#define WS_K     (WS_Q     + (size_t)NN*Dm*2)
#define WS_V     (WS_K     + (size_t)NN*Dm*2)
#define WS_BP    (WS_V     + (size_t)NN*Dm*2)                 // [h][i][j] f32
#define WS_GATE  (WS_BP    + (size_t)Hh*NN*4)                 // [g][128] f16 (post-sigmoid)
#define WS_AOUT  (WS_GATE  + (size_t)NN*Dm*2)                 // [g][128] f16 (gated attn out)

// ---- WMMA fragment helpers (contiguous k order k = half*16 + e, same for A and B) ----
union AFrag { v16h v; uint4 u[2]; _Float16 h[16]; };

// A-matrix 16x32 f16: lane M=lane%16; elems e = K[half*16 + e]
__device__ __forceinline__ v16h load_afrag(const _Float16* row, int half, int kbase) {
  AFrag f;
  f.u[0] = *reinterpret_cast<const uint4*>(row + kbase + half*16);
  f.u[1] = *reinterpret_cast<const uint4*>(row + kbase + half*16 + 8);
  return f.v;
}
// B-matrix 32x16 f16: lane N=lane%16; elems e = K[half*16 + e]  (32 contiguous bytes)
__device__ __forceinline__ v16h load_bfrag(const _Float16* p) {
  AFrag f;
  f.u[0] = *reinterpret_cast<const uint4*>(p);
  f.u[1] = *reinterpret_cast<const uint4*>(p + 8);
  return f.v;
}
__device__ __forceinline__ v8f wmma_f16(v16h a, v16h b, v8f c) {
  return __builtin_amdgcn_wmma_f32_16x16x32_f16(false, a, false, b, (short)0, c, false, false);
}
// fast reciprocal: single v_rcp_f32, no IEEE div fixup sequence
__device__ __forceinline__ float fast_rcp(float x) { return __builtin_amdgcn_rcpf(x); }

// ================= Kernel 0: weight convert + transpose =================
__global__ __launch_bounds__(256) void prep_weights(
    const float* __restrict__ w_qkv, const float* __restrict__ w_pair,
    const float* __restrict__ w_gate, const float* __restrict__ w_out,
    _Float16* __restrict__ w1t, _Float16* __restrict__ woutt)
{
  int idx = blockIdx.x * 256 + threadIdx.x;
  if (idx < W1COLS * Dm) {               // w1t[col][k]
    int col = idx / Dm, k = idx % Dm;
    float v = 0.f;
    if      (col < 384) v = w_qkv[k * 384 + col];
    else if (col < 512) v = w_gate[k * Dm + (col - 384)];
    else if (col < 516) v = w_pair[k * 4  + (col - 512)];
    w1t[idx] = (_Float16)v;
  }
  idx -= W1COLS * Dm;
  if (idx >= 0 && idx < Dm * Dm) {       // woutt[c][k] = w_out[k][c]
    int c = idx / Dm, k = idx % Dm;
    woutt[idx] = (_Float16)w_out[k * Dm + c];
  }
}

// ============ Kernel 1: LayerNorm + fused qkv/gate/pair projection ============
__global__ __launch_bounds__(256) void ln_project(
    const float* __restrict__ z, const float* __restrict__ ln_scale,
    const float* __restrict__ ln_bias, const float* __restrict__ b_gate,
    const _Float16* __restrict__ w1t,
    _Float16* __restrict__ qws, _Float16* __restrict__ kws, _Float16* __restrict__ vws,
    float* __restrict__ bpair, _Float16* __restrict__ gate)
{
  __shared__ __align__(16) _Float16 zn[16][Dm];   // 4 KB
  const int tid = threadIdx.x;
  const int g0  = blockIdx.x * 16;                // 16 positions per block

  // ---- LayerNorm: 16 lanes per row (half-wave), 8 elems per lane ----
  {
    const int row = tid >> 4, sub = tid & 15;
    const float* zp = z + (size_t)(g0 + row) * Dm + sub * 8;
    float4 p0 = *reinterpret_cast<const float4*>(zp);
    float4 p1 = *reinterpret_cast<const float4*>(zp + 4);
    float x[8] = {p0.x, p0.y, p0.z, p0.w, p1.x, p1.y, p1.z, p1.w};
    float s = 0.f, ss = 0.f;
    #pragma unroll
    for (int e = 0; e < 8; e++) { s += x[e]; ss += x[e] * x[e]; }
    #pragma unroll
    for (int m = 1; m <= 8; m <<= 1) { s += __shfl_xor(s, m); ss += __shfl_xor(ss, m); }
    const float mu   = s * (1.f / 128.f);
    const float var  = ss * (1.f / 128.f) - mu * mu;
    const float rstd = rsqrtf(var + LN_EPS);
    #pragma unroll
    for (int e = 0; e < 8; e++) {
      int d = sub * 8 + e;
      zn[row][d] = (_Float16)((x[e] - mu) * rstd * ln_scale[d] + ln_bias[d]);
    }
  }
  __syncthreads();

  // ---- zn[16x128] @ W1[128x528] via 4 chained WMMAs per 16-wide col tile ----
  const int lane = tid & 31, wv = tid >> 5;
  const int m16 = lane & 15, half = lane >> 4;
  v16h a[4];
  #pragma unroll
  for (int kt = 0; kt < 4; kt++) a[kt] = load_afrag(&zn[m16][0], half, kt * 32);

  for (int t = wv; t < 33; t += 8) {
    v8f c = {};
    #pragma unroll
    for (int kt = 0; kt < 4; kt++) {
      v16h b = load_bfrag(w1t + (size_t)(t * 16 + m16) * Dm + kt * 32 + half * 16);
      c = wmma_f16(a[kt], b, c);
    }
    // epilogue scatter: C lane holds (M = v + half*8, N = m16)
    #pragma unroll
    for (int vv = 0; vv < 8; vv++) {
      const int il  = vv + half * 8;
      const int g   = g0 + il;
      const int col = t * 16 + m16;
      const float val = c[vv];
      if (col < 384) {                       // q/k/v, head-major (h d)
        const int which = col >> 7;
        const int h = (col & 127) >> 5, d = col & 31;
        const int r = g >> 8, i = g & 255;
        const size_t o = ((size_t)(r * Hh + h) * Nseq + i) * DHd + d;
        _Float16* p = (which == 0) ? qws : (which == 1) ? kws : vws;
        p[o] = (_Float16)val;
      } else if (col < 512) {                // gate = sigmoid(x + b_gate), fast rcp
        const int d = col - 384;
        const float gv = val + b_gate[d];
        gate[(size_t)g * Dm + d] = (_Float16)fast_rcp(1.f + __expf(-gv));
      } else if (col < 516) {                // pair bias, [h][i][j]
        bpair[(size_t)(col - 512) * NN + g] = val;
      }
    }
  }
}

// ================= Kernel 2: row-wise attention per (r, h, 16 queries) =================
__global__ __launch_bounds__(64) void attention(
    const _Float16* __restrict__ qws, const _Float16* __restrict__ kws,
    const _Float16* __restrict__ vws, const float* __restrict__ bpair,
    const _Float16* __restrict__ gate, const int* __restrict__ src_mask,
    _Float16* __restrict__ aout)
{
  __shared__ __align__(16) _Float16 kf[Nseq][DHd];     // 16 KB, row-major [j][d]
  __shared__ __align__(16) _Float16 vT[DHd][Nseq];     // 16 KB, transposed [d][j]
  __shared__ __align__(16) float    logits[16][Nseq];  // 16 KB
  __shared__ __align__(16) _Float16 attn[16][Nseq];    //  8 KB (unnormalized exp)
  __shared__ float mf[Nseq];
  __shared__ float red[16][4];
  __shared__ float rowmax[16];
  __shared__ float rinv[16];

  const int tid = threadIdx.x;
  const int i0 = blockIdx.x * 16, h = blockIdx.y, r = blockIdx.z;
  const size_t kvbase = (size_t)(r * Hh + h) * Nseq * DHd;

  // ---- stage 0: stage K (row-major) and V (transposed) into LDS ----
  for (int idx = tid; idx < Nseq * DHd / 8; idx += 64) {
    const int j = idx >> 2, d0 = (idx & 3) * 8;
    *reinterpret_cast<uint4*>(&kf[j][d0]) =
        *reinterpret_cast<const uint4*>(kws + kvbase + (size_t)j * DHd + d0);
    AFrag vv;
    vv.u[0] = *reinterpret_cast<const uint4*>(vws + kvbase + (size_t)j * DHd + d0);
    #pragma unroll
    for (int e = 0; e < 8; e++) vT[d0 + e][j] = vv.h[e];
  }
  for (int idx = tid; idx < Nseq; idx += 64) mf[idx] = src_mask[idx] ? 1.f : -1.f;
  __syncthreads();

  const int lane = tid & 31, wv = tid >> 5, m16 = lane & 15, half = lane >> 4;
  const _Float16* qrow = qws + kvbase + (size_t)(i0 + m16) * DHd;
  const v16h aq = load_afrag(qrow, half, 0);          // 16x32 query tile (K = DH = 32)
  const float rscale = 0.17677669529663689f;          // 1/sqrt(32)

  // ---- stage 1: logits = q k^T / sqrt(DH) + bpair, faithful -1e-9 mask ----
  for (int jt = wv; jt < 16; jt += 2) {
    v16h b = load_bfrag(&kf[jt * 16 + m16][half * 16]);
    v8f c = {};
    c = wmma_f16(aq, b, c);
    const int j = jt * 16 + m16;
    const float mj = mf[j];
    #pragma unroll
    for (int vv = 0; vv < 8; vv++) {
      const int il = vv + half * 8, ig = i0 + il;
      float val = c[vv] * rscale + bpair[(size_t)h * NN + (size_t)ig * Nseq + j];
      if (mf[ig] * mj < 0.f) val = -1e-9f;           // reference keeps the -1e-9 "bug"
      logits[il][j] = val;
    }
  }
  __syncthreads();

  // ---- stage 2: softmax stats (4 threads per row, 64 cols each) ----
  {
    const int rr = tid & 15, ch = tid >> 4;
    float mx = -3.0e38f;
    for (int j = ch * 64; j < ch * 64 + 64; j++) mx = fmaxf(mx, logits[rr][j]);
    red[rr][ch] = mx;
  }
  __syncthreads();
  if (tid < 16)
    rowmax[tid] = fmaxf(fmaxf(red[tid][0], red[tid][1]), fmaxf(red[tid][2], red[tid][3]));
  __syncthreads();
  {
    const int rr = tid & 15, ch = tid >> 4;
    const float mx = rowmax[rr];
    float sm = 0.f;
    for (int j = ch * 64; j < ch * 64 + 64; j++) {
      const float e = __expf(logits[rr][j] - mx);
      attn[rr][j] = (_Float16)e;                     // unnormalized; 1/sum folded later
      sm += e;
    }
    red[rr][ch] = sm;
  }
  __syncthreads();
  if (tid < 16)
    rinv[tid] = fast_rcp(red[tid][0] + red[tid][1] + red[tid][2] + red[tid][3]);
  __syncthreads();

  // ---- stage 3: out = attn @ V (wave wv owns d-tile wv), fold 1/sum and gate ----
  const int d0 = wv * 16;
  v8f c = {};
  #pragma unroll
  for (int jc = 0; jc < 8; jc++) {
    v16h aa = load_afrag(&attn[m16][0], half, jc * 32);
    v16h bb = load_bfrag(&vT[d0 + m16][jc * 32 + half * 16]);
    c = wmma_f16(aa, bb, c);
  }
  #pragma unroll
  for (int vv = 0; vv < 8; vv++) {
    const int il = vv + half * 8, ig = i0 + il;
    const int d = d0 + m16;
    const size_t gi = ((size_t)(r * Nseq + ig)) * Dm + h * DHd + d;
    aout[gi] = (_Float16)(c[vv] * rinv[il] * (float)gate[gi]);
  }
}

// ================= Kernel 3: output projection + bias =================
__global__ __launch_bounds__(256) void out_proj(
    const _Float16* __restrict__ aout, const _Float16* __restrict__ woutt,
    const float* __restrict__ b_out, float* __restrict__ out)
{
  const int tid = threadIdx.x;
  const int g0 = blockIdx.x * 16;
  const int lane = tid & 31, wv = tid >> 5, m16 = lane & 15, half = lane >> 4;
  const _Float16* arow = aout + (size_t)(g0 + m16) * Dm;
  v8f c = {};
  #pragma unroll
  for (int kt = 0; kt < 4; kt++) {
    v16h a = load_afrag(arow, half, kt * 32);
    v16h b = load_bfrag(woutt + (size_t)(wv * 16 + m16) * Dm + kt * 32 + half * 16);
    c = wmma_f16(a, b, c);
  }
  const float bo = b_out[wv * 16 + m16];
  #pragma unroll
  for (int vv = 0; vv < 8; vv++)
    out[(size_t)(g0 + vv + half * 8) * Dm + wv * 16 + m16] = c[vv] + bo;
}

// ================= launch =================
extern "C" void kernel_launch(void* const* d_in, const int* in_sizes, int n_in,
                              void* d_out, int out_size, void* d_ws, size_t ws_size,
                              hipStream_t stream) {
  (void)in_sizes; (void)n_in; (void)out_size; (void)ws_size;
  const float* z        = (const float*)d_in[0];
  const int*   src_mask = (const int*)  d_in[1];
  const float* ln_scale = (const float*)d_in[2];
  const float* ln_bias  = (const float*)d_in[3];
  const float* w_qkv    = (const float*)d_in[4];
  const float* w_pair   = (const float*)d_in[5];
  const float* w_gate   = (const float*)d_in[6];
  const float* b_gate   = (const float*)d_in[7];
  const float* w_out    = (const float*)d_in[8];
  const float* b_out    = (const float*)d_in[9];
  float* out = (float*)d_out;

  char* ws = (char*)d_ws;
  _Float16* w1t   = (_Float16*)(ws + WS_W1T);
  _Float16* woutt = (_Float16*)(ws + WS_WOUTT);
  _Float16* qws   = (_Float16*)(ws + WS_Q);
  _Float16* kws   = (_Float16*)(ws + WS_K);
  _Float16* vws   = (_Float16*)(ws + WS_V);
  float*    bp    = (float*)   (ws + WS_BP);
  _Float16* gate  = (_Float16*)(ws + WS_GATE);
  _Float16* aoutb = (_Float16*)(ws + WS_AOUT);

  prep_weights<<<(W1COLS * Dm + Dm * Dm + 255) / 256, 256, 0, stream>>>(
      w_qkv, w_pair, w_gate, w_out, w1t, woutt);

  ln_project<<<NN / 16, 256, 0, stream>>>(
      z, ln_scale, ln_bias, b_gate, w1t, qws, kws, vws, bp, gate);

  dim3 gb(Nseq / 16, Hh, Nseq);
  attention<<<gb, 64, 0, stream>>>(qws, kws, vws, bp, gate, src_mask, aoutb);

  out_proj<<<NN / 16, 256, 0, stream>>>(aoutb, woutt, b_out, out);
}
